// AttentionalGNN_274877907595
// MI455X (gfx1250) — matrix operand order repaired and checked by reading, not compiled, not run on hardware
//
#include <hip/hip_runtime.h>
#include <math.h>

// ---------------------------------------------------------------------------
// CDNA5 (gfx1250) wave32 WMMA types & helpers
// ---------------------------------------------------------------------------
typedef __attribute__((ext_vector_type(16))) _Float16 v16h;
typedef __attribute__((ext_vector_type(8)))  float    v8f;

#define NNODES 4096
#define DFEAT  256
#define NSUP   128
#define BN_EPS 1e-5f

__device__ __forceinline__ v8f wmma_f32_16x16x32_f16(v16h a, v16h b, v8f c) {
  // (neg_a, A, neg_b, B, c_mod, C, reuse_a, reuse_b)
  return __builtin_amdgcn_wmma_f32_16x16x32_f16(false, a, false, b, (short)0, c,
                                                false, false);
}

// A operand (16x32 f16, M x K): lane L holds row M=L%16.
// Per ISA 7.12.2: lanes 0-15: VGPR0..3 -> K=0..7 (pairs), VGPR4..7 -> K=16..23;
// lanes 16-31 add +8 to K. LDS tile is [row][k] with k contiguous
// (lowers to two ds_load_b128 per fragment).
__device__ __forceinline__ v16h load_a_frag(const _Float16* base, int row,
                                            int stride, int kk, int lane) {
  const _Float16* p = base + row * stride + kk + ((lane & 16) ? 8 : 0);
  v16h a;
#pragma unroll
  for (int v = 0; v < 8; ++v) {
    int k = ((v & 4) << 2) + ((v & 3) << 1);  // {0,2,4,6,16,18,20,22}
    a[2 * v]     = p[k];
    a[2 * v + 1] = p[k + 1];
  }
  return a;
}

// B operand (32x16 f16, K x N): lane L holds column N=L%16; lanes 0-15 hold
// K=0..7,16..23, lanes 16-31 hold K=8..15,24..31. LDS tile is [col][k], k contiguous.
__device__ __forceinline__ v16h load_b_frag(const _Float16* base, int col,
                                            int stride, int kk, int lane) {
  const _Float16* p = base + col * stride + kk + ((lane & 16) ? 8 : 0);
  v16h b;
#pragma unroll
  for (int i = 0; i < 16; ++i) b[i] = p[i < 8 ? i : i + 8];
  return b;
}

// ---------------------------------------------------------------------------
// sklearn preprocessing.scale on (Nr, C) then transpose -> out (C, Nr)
// one block per column c
// ---------------------------------------------------------------------------
__global__ __launch_bounds__(256) void scale_transpose_kernel(
    const float* __restrict__ P, float* __restrict__ O, int Nr, int C) {
  __shared__ float red[256];
  const int c = blockIdx.x;
  const int tid = threadIdx.x;
  float s = 0.f, s2 = 0.f;
  for (int n = tid; n < Nr; n += 256) {
    float v = P[(long)n * C + c];
    s += v;
    s2 += v * v;
  }
  red[tid] = s;
  __syncthreads();
  for (int o = 128; o > 0; o >>= 1) {
    if (tid < o) red[tid] += red[tid + o];
    __syncthreads();
  }
  const float mean = red[0] / (float)Nr;
  __syncthreads();
  red[tid] = s2;
  __syncthreads();
  for (int o = 128; o > 0; o >>= 1) {
    if (tid < o) red[tid] += red[tid + o];
    __syncthreads();
  }
  float var = red[0] / (float)Nr - mean * mean;
  float sd = sqrtf(fmaxf(var, 0.f));
  float inv = (sd == 0.f) ? 1.f : 1.f / sd;  // zero-std columns left as-is
  for (int n = tid; n < Nr; n += 256)
    O[(long)c * Nr + n] = (P[(long)n * C + c] - mean) * inv;
}

// ---------------------------------------------------------------------------
// Y(Cout,Nn) = W(Cout,Cin) @ X(Cin,Nn) + bias   via v_wmma_f32_16x16x32_f16
// block = 256 threads = 8 waves, 64x64 output tile, K stepped by 32
// ---------------------------------------------------------------------------
__global__ __launch_bounds__(256) void gemm_bias_kernel(
    const float* __restrict__ W, const float* __restrict__ X,
    const float* __restrict__ bias, float* __restrict__ Y,
    int Cout, int Cin, int Nn) {
  __shared__ _Float16 At[64 * 32];  // [m][k]
  __shared__ _Float16 Bt[64 * 32];  // [n][k]
  const int tid  = threadIdx.x;
  const int lane = tid & 31;
  const int w    = tid >> 5;
  const int n0   = blockIdx.x * 64;
  const int m0   = blockIdx.y * 64;
  const int rw   = (w & 3) * 16;   // wave's M rows within tile
  const int cw   = (w >> 2) * 32;  // wave's N cols within tile
  v8f c0 = {};
  v8f c1 = {};
  for (int k0 = 0; k0 < Cin; k0 += 32) {
    __syncthreads();
#pragma unroll
    for (int i = 0; i < 8; ++i) {
      int idx = tid + i * 256;
      int k = idx & 31, m = idx >> 5;
      At[m * 32 + k] = (_Float16)W[(long)(m0 + m) * Cin + k0 + k];
    }
#pragma unroll
    for (int i = 0; i < 8; ++i) {
      int idx = tid + i * 256;
      int n = idx & 63, k = idx >> 6;
      Bt[n * 32 + k] = (_Float16)X[(long)(k0 + k) * Nn + n0 + n];
    }
    __syncthreads();
    v16h a  = load_a_frag(At, rw + (lane & 15), 32, 0, lane);
    v16h b0 = load_b_frag(Bt, cw + (lane & 15), 32, 0, lane);
    v16h b1 = load_b_frag(Bt, cw + 16 + (lane & 15), 32, 0, lane);
    c0 = wmma_f32_16x16x32_f16(a, b0, c0);
    c1 = wmma_f32_16x16x32_f16(a, b1, c1);
  }
  const int col   = n0 + cw + (lane & 15);
  const int rbase = m0 + rw + ((lane & 16) ? 8 : 0);
#pragma unroll
  for (int j = 0; j < 8; ++j) {
    float bv = bias[rbase + j];
    Y[(long)(rbase + j) * Nn + col]      = c0[j] + bv;
    Y[(long)(rbase + j) * Nn + col + 16] = c1[j] + bv;
  }
}

// ---------------------------------------------------------------------------
// In-place BatchNorm1d (training batch stats over Nn) + ReLU; block per channel
// ---------------------------------------------------------------------------
__global__ __launch_bounds__(256) void bn_relu_kernel(
    float* __restrict__ X, const float* __restrict__ g,
    const float* __restrict__ be, int Nn) {
  __shared__ float red[256];
  const int c = blockIdx.x;
  const int tid = threadIdx.x;
  float s = 0.f, s2 = 0.f;
  for (int n = tid; n < Nn; n += 256) {
    float v = X[(long)c * Nn + n];
    s += v;
    s2 += v * v;
  }
  red[tid] = s;
  __syncthreads();
  for (int o = 128; o > 0; o >>= 1) {
    if (tid < o) red[tid] += red[tid + o];
    __syncthreads();
  }
  const float mean = red[0] / (float)Nn;
  __syncthreads();
  red[tid] = s2;
  __syncthreads();
  for (int o = 128; o > 0; o >>= 1) {
    if (tid < o) red[tid] += red[tid + o];
    __syncthreads();
  }
  const float var  = red[0] / (float)Nn - mean * mean;
  const float inv  = rsqrtf(var + BN_EPS);
  const float sc   = g[c] * inv;
  const float sh   = be[c] - mean * sc;
  for (int n = tid; n < Nn; n += 256) {
    float v = X[(long)c * Nn + n] * sc + sh;
    X[(long)c * Nn + n] = fmaxf(v, 0.f);
  }
}

// ---------------------------------------------------------------------------
// Fused flash attention, SINGLE-PASS online softmax (no NxN materialization,
// each QK^T tile computed exactly once; persistent WMMA accumulators rescaled
// by alpha = exp(m_old - m_new) per key tile).
// Head h uses channels d = dd*4 + h (SuperGlue reshape (D,N)->(HD,H,N)).
// Block = (query tile of 64, head); grid = (Nn/64, 4).
// ---------------------------------------------------------------------------
__global__ __launch_bounds__(256) void flash_attn_kernel(
    const float* __restrict__ Qg, const float* __restrict__ Kg,
    const float* __restrict__ Vg, float* __restrict__ Og, int Nn) {
  __shared__ _Float16 Aq[64 * 64];  // [query][dd]   (A for S)
  __shared__ _Float16 Bk[64 * 64];  // [key][dd]     (B for S)
  __shared__ _Float16 Vt[64 * 64];  // [dd][key]     (B for O)
  __shared__ _Float16 Pt[64 * 64];  // [query][key]  (A for O)
  __shared__ float    Sb[64 * 64];  // score tile f32
  __shared__ float    rowmax[64];   // running m_i
  __shared__ float    rowsum[64];   // running l_i
  __shared__ float    alpha[64];    // exp(m_old - m_new) for current tile
  __shared__ float    pred[4 * 64]; // 4-way partial max/sum staging
  const int tid  = threadIdx.x;
  const int lane = tid & 31;
  const int w    = tid >> 5;
  const int h    = blockIdx.y;
  const int n0   = blockIdx.x * 64;
  const int rw   = (w & 3) * 16;
  const int cw   = (w >> 2) * 32;
  const int seg  = tid >> 6;  // 0..3 : 16-key segment for softmax work
  const int row  = tid & 63;  // query row for softmax work

  // Q tile, with 1/sqrt(HD)=0.125 folded in
#pragma unroll
  for (int i = 0; i < 16; ++i) {
    int idx = tid + i * 256;
    int n = idx & 63, dd = idx >> 6;
    Aq[n * 64 + dd] = (_Float16)(0.125f * Qg[(long)(dd * 4 + h) * Nn + n0 + n]);
  }
  if (tid < 64) {
    rowmax[tid] = -1e30f;
    rowsum[tid] = 0.f;
  }
  v8f o0 = {};
  v8f o1 = {};

  for (int m0 = 0; m0 < Nn; m0 += 64) {
    __syncthreads();  // previous-iteration consumers of Bk/Vt/Pt/alpha done
    // ---- load K & V tiles ----
#pragma unroll
    for (int i = 0; i < 16; ++i) {
      int idx = tid + i * 256;
      int m = idx & 63, dd = idx >> 6;
      Bk[m * 64 + dd] = (_Float16)Kg[(long)(dd * 4 + h) * Nn + m0 + m];
      Vt[dd * 64 + m] = (_Float16)Vg[(long)(dd * 4 + h) * Nn + m0 + m];
    }
    __syncthreads();
    // ---- S = (Q/8)^T K over this tile ----
    v8f s0 = {};
    v8f s1 = {};
#pragma unroll
    for (int kk = 0; kk < 64; kk += 32) {
      v16h a  = load_a_frag(Aq, rw + (lane & 15), 64, kk, lane);
      v16h b0 = load_b_frag(Bk, cw + (lane & 15), 64, kk, lane);
      v16h b1 = load_b_frag(Bk, cw + 16 + (lane & 15), 64, kk, lane);
      s0 = wmma_f32_16x16x32_f16(a, b0, s0);
      s1 = wmma_f32_16x16x32_f16(a, b1, s1);
    }
    {
      int col = cw + (lane & 15);
      int rb  = rw + ((lane & 16) ? 8 : 0);
#pragma unroll
      for (int j = 0; j < 8; ++j) {
        Sb[(rb + j) * 64 + col]      = s0[j];
        Sb[(rb + j) * 64 + col + 16] = s1[j];
      }
    }
    __syncthreads();
    // ---- online softmax: new row max (all 256 threads) ----
    {
      const float* srow = &Sb[row * 64 + seg * 16];
      float pm = -1e30f;
#pragma unroll
      for (int q = 0; q < 16; ++q) pm = fmaxf(pm, srow[q]);
      pred[seg * 64 + row] = pm;
    }
    __syncthreads();
    if (tid < 64) {
      float om = rowmax[tid];
      float nm = fmaxf(fmaxf(pred[tid], pred[64 + tid]),
                       fmaxf(pred[128 + tid], pred[192 + tid]));
      nm = fmaxf(nm, om);
      float al = __expf(om - nm);  // first tile: exp(-inf) == 0
      alpha[tid]  = al;
      rowmax[tid] = nm;
      rowsum[tid] *= al;
    }
    __syncthreads();
    // ---- P = exp(S - m_new), partial row sums (all 256 threads) ----
    {
      const float rm = rowmax[row];
      const float* srow = &Sb[row * 64 + seg * 16];
      _Float16* prow = &Pt[row * 64 + seg * 16];
      float ps = 0.f;
#pragma unroll
      for (int q = 0; q < 16; ++q) {
        float e = __expf(srow[q] - rm);
        prow[q] = (_Float16)e;
        ps += e;
      }
      pred[seg * 64 + row] = ps;
    }
    __syncthreads();
    if (tid < 64)
      rowsum[tid] += pred[tid] + pred[64 + tid] + pred[128 + tid] + pred[192 + tid];
    // ---- rescale accumulators, O += P @ V^T ----
    {
      int rb = rw + ((lane & 16) ? 8 : 0);
#pragma unroll
      for (int j = 0; j < 8; ++j) {
        float al = alpha[rb + j];
        o0[j] *= al;
        o1[j] *= al;
      }
    }
#pragma unroll
    for (int kk = 0; kk < 64; kk += 32) {
      v16h a  = load_a_frag(Pt, rw + (lane & 15), 64, kk, lane);
      v16h b0 = load_b_frag(Vt, cw + (lane & 15), 64, kk, lane);
      v16h b1 = load_b_frag(Vt, cw + 16 + (lane & 15), 64, kk, lane);
      o0 = wmma_f32_16x16x32_f16(a, b0, o0);
      o1 = wmma_f32_16x16x32_f16(a, b1, o1);
    }
  }
  __syncthreads();  // rowsum final

  // normalize + scatter back to (D, N) with d = dd*4 + h
  const int dd0 = cw + (lane & 15);
  const int rb  = rw + ((lane & 16) ? 8 : 0);
#pragma unroll
  for (int j = 0; j < 8; ++j) {
    float inv = 1.f / rowsum[rb + j];
    Og[(long)(dd0 * 4 + h) * Nn + n0 + rb + j]        = o0[j] * inv;
    Og[(long)((dd0 + 16) * 4 + h) * Nn + n0 + rb + j] = o1[j] * inv;
  }
}

// ---------------------------------------------------------------------------
// elementwise helpers
// ---------------------------------------------------------------------------
__global__ void mul_kernel(const float* __restrict__ a,
                           const float* __restrict__ b,
                           float* __restrict__ z, long n) {
  for (long i = (long)blockIdx.x * 256 + threadIdx.x; i < n;
       i += (long)gridDim.x * 256)
    z[i] = a[i] * b[i];
}

__global__ void copy_kernel(const float* __restrict__ a, float* __restrict__ z,
                            long n) {
  for (long i = (long)blockIdx.x * 256 + threadIdx.x; i < n;
       i += (long)gridDim.x * 256)
    z[i] = a[i];
}

// ---------------------------------------------------------------------------
// out = || mean_n(xs + y2s) - mean_n(xt + y2t) ||^2   (one block, 256 ch)
// ---------------------------------------------------------------------------
__global__ __launch_bounds__(256) void final_mmd_kernel(
    const float* __restrict__ xs, const float* __restrict__ y2s,
    const float* __restrict__ xt, const float* __restrict__ y2t,
    float* __restrict__ out, int Nn) {
  __shared__ float red[256];
  const int c = threadIdx.x;
  float acc = 0.f;
  for (int n = 0; n < Nn; ++n)
    acc += (xs[(long)c * Nn + n] + y2s[(long)c * Nn + n]) -
           (xt[(long)c * Nn + n] + y2t[(long)c * Nn + n]);
  float d = acc / (float)Nn;
  red[c] = d * d;
  __syncthreads();
  for (int o = 128; o > 0; o >>= 1) {
    if (c < o) red[c] += red[c + o];
    __syncthreads();
  }
  if (c == 0) out[0] = red[0];
}

// ---------------------------------------------------------------------------
// host orchestration
// ---------------------------------------------------------------------------
extern "C" void kernel_launch(void* const* d_in, const int* in_sizes, int n_in,
                              void* d_out, int out_size, void* d_ws,
                              size_t ws_size, hipStream_t stream) {
  (void)in_sizes; (void)n_in; (void)out_size; (void)ws_size;
  const float* p_src  = (const float*)d_in[0];
  const float* p_tar  = (const float*)d_in[1];
  const float* dis_s  = (const float*)d_in[2];
  const float* dis_t  = (const float*)d_in[3];
  const float* aq_w = (const float*)d_in[4];  const float* aq_b = (const float*)d_in[5];
  const float* ak_w = (const float*)d_in[6];  const float* ak_b = (const float*)d_in[7];
  const float* av_w = (const float*)d_in[8];  const float* av_b = (const float*)d_in[9];
  const float* am_w = (const float*)d_in[10]; const float* am_b = (const float*)d_in[11];
  const float* m1_w = (const float*)d_in[12]; const float* m1_b = (const float*)d_in[13];
  const float* m1_g = (const float*)d_in[14]; const float* m1_be = (const float*)d_in[15];
  const float* m2_w = (const float*)d_in[16]; const float* m2_b = (const float*)d_in[17];
  const float* d1_w = (const float*)d_in[18]; const float* d1_b = (const float*)d_in[19];
  const float* d1_g = (const float*)d_in[20]; const float* d1_be = (const float*)d_in[21];
  const float* d2_w = (const float*)d_in[22]; const float* d2_b = (const float*)d_in[23];
  float* out = (float*)d_out;

  const long U = (long)DFEAT * NNODES;  // 1,048,576 floats = 4MB
  float* ws = (float*)d_ws;
  float* xs = ws + 0 * U;
  float* xt = ws + 1 * U;
  float* ds = ws + 2 * U;
  float* dt = ws + 3 * U;
  float* Qb = ws + 4 * U;
  float* Kb = ws + 5 * U;
  float* Vb = ws + 6 * U;
  float* xO = ws + 7 * U;
  float* dl0 = ws + 8 * U;
  float* dl1 = ws + 9 * U;
  float* pp = ws + 10 * U;
  float* pt = ws + 11 * U;
  // lifetime-safe aliases (total footprint 12*4MB = 48MB, fits in L2):
  float* scd  = Vb;   // scaled dis (128 x 4096), before attention phase
  float* hdis = Qb;   // mlp_dis hidden (256 x 4096), before attention phase
  float* hmlp = Qb;   // mlp hidden (512 x 4096) spans Qb..Kb, after attention
  float* cat  = pp;   // concat input (512 x 4096) spans pp..pt, after gating use
  float* y2s  = xO;   // mlp output src (after last attention merge)
  float* y2t  = dl1;  // mlp output tar (after cat_t copies dl1)

  const dim3 B(256);
  const dim3 G64x4(NNODES / 64, DFEAT / 64);        // Cout=256
  const dim3 G64x8(NNODES / 64, (2 * DFEAT) / 64);  // Cout=512
  const dim3 GF(NNODES / 64, 4);                    // flash: (query tiles, heads)

  // --- scale ---
  scale_transpose_kernel<<<DFEAT, B, 0, stream>>>(p_src, xs, NNODES, DFEAT);
  scale_transpose_kernel<<<DFEAT, B, 0, stream>>>(p_tar, xt, NNODES, DFEAT);

  // --- mlp_dis (src) ---
  scale_transpose_kernel<<<NSUP, B, 0, stream>>>(dis_s, scd, NNODES, NSUP);
  gemm_bias_kernel<<<G64x4, B, 0, stream>>>(d1_w, scd, d1_b, hdis, DFEAT, NSUP, NNODES);
  bn_relu_kernel<<<DFEAT, B, 0, stream>>>(hdis, d1_g, d1_be, NNODES);
  gemm_bias_kernel<<<G64x4, B, 0, stream>>>(d2_w, hdis, d2_b, ds, DFEAT, DFEAT, NNODES);
  // --- mlp_dis (tar) ---
  scale_transpose_kernel<<<NSUP, B, 0, stream>>>(dis_t, scd, NNODES, NSUP);
  gemm_bias_kernel<<<G64x4, B, 0, stream>>>(d1_w, scd, d1_b, hdis, DFEAT, NSUP, NNODES);
  bn_relu_kernel<<<DFEAT, B, 0, stream>>>(hdis, d1_g, d1_be, NNODES);
  gemm_bias_kernel<<<G64x4, B, 0, stream>>>(d2_w, hdis, d2_b, dt, DFEAT, DFEAT, NNODES);

  // --- attention 1: attn(xs, xt, xt) -> dl0 ---
  gemm_bias_kernel<<<G64x4, B, 0, stream>>>(aq_w, xs, aq_b, Qb, DFEAT, DFEAT, NNODES);
  gemm_bias_kernel<<<G64x4, B, 0, stream>>>(ak_w, xt, ak_b, Kb, DFEAT, DFEAT, NNODES);
  gemm_bias_kernel<<<G64x4, B, 0, stream>>>(av_w, xt, av_b, Vb, DFEAT, DFEAT, NNODES);
  flash_attn_kernel<<<GF, B, 0, stream>>>(Qb, Kb, Vb, xO, NNODES);
  gemm_bias_kernel<<<G64x4, B, 0, stream>>>(am_w, xO, am_b, dl0, DFEAT, DFEAT, NNODES);
  // --- attention 2: attn(xt, xs, xs) -> dl1 ---
  gemm_bias_kernel<<<G64x4, B, 0, stream>>>(aq_w, xt, aq_b, Qb, DFEAT, DFEAT, NNODES);
  gemm_bias_kernel<<<G64x4, B, 0, stream>>>(ak_w, xs, ak_b, Kb, DFEAT, DFEAT, NNODES);
  gemm_bias_kernel<<<G64x4, B, 0, stream>>>(av_w, xs, av_b, Vb, DFEAT, DFEAT, NNODES);
  flash_attn_kernel<<<GF, B, 0, stream>>>(Qb, Kb, Vb, xO, NNODES);
  gemm_bias_kernel<<<G64x4, B, 0, stream>>>(am_w, xO, am_b, dl1, DFEAT, DFEAT, NNODES);

  // --- gating: ps_tmp = dl0 * xt ; pt_tmp = dl1 * xs ---
  mul_kernel<<<4096, B, 0, stream>>>(dl0, xt, pp, U);
  mul_kernel<<<4096, B, 0, stream>>>(dl1, xs, pt, U);

  // --- attention 3: attn(ds, dt, ps_tmp) -> dl0 ---
  gemm_bias_kernel<<<G64x4, B, 0, stream>>>(aq_w, ds, aq_b, Qb, DFEAT, DFEAT, NNODES);
  gemm_bias_kernel<<<G64x4, B, 0, stream>>>(ak_w, dt, ak_b, Kb, DFEAT, DFEAT, NNODES);
  gemm_bias_kernel<<<G64x4, B, 0, stream>>>(av_w, pp, av_b, Vb, DFEAT, DFEAT, NNODES);
  flash_attn_kernel<<<GF, B, 0, stream>>>(Qb, Kb, Vb, xO, NNODES);
  gemm_bias_kernel<<<G64x4, B, 0, stream>>>(am_w, xO, am_b, dl0, DFEAT, DFEAT, NNODES);
  // --- attention 4: attn(dt, ds, pt_tmp) -> dl1 ---
  gemm_bias_kernel<<<G64x4, B, 0, stream>>>(aq_w, dt, aq_b, Qb, DFEAT, DFEAT, NNODES);
  gemm_bias_kernel<<<G64x4, B, 0, stream>>>(ak_w, ds, ak_b, Kb, DFEAT, DFEAT, NNODES);
  gemm_bias_kernel<<<G64x4, B, 0, stream>>>(av_w, pt, av_b, Vb, DFEAT, DFEAT, NNODES);
  flash_attn_kernel<<<GF, B, 0, stream>>>(Qb, Kb, Vb, xO, NNODES);
  gemm_bias_kernel<<<G64x4, B, 0, stream>>>(am_w, xO, am_b, dl1, DFEAT, DFEAT, NNODES);

  // --- mlp (src): cat=[xs; dl0] -> conv -> BN/ReLU -> conv -> y2s ---
  copy_kernel<<<4096, B, 0, stream>>>(xs, cat, U);
  copy_kernel<<<4096, B, 0, stream>>>(dl0, cat + U, U);
  gemm_bias_kernel<<<G64x8, B, 0, stream>>>(m1_w, cat, m1_b, hmlp, 2 * DFEAT, 2 * DFEAT, NNODES);
  bn_relu_kernel<<<2 * DFEAT, B, 0, stream>>>(hmlp, m1_g, m1_be, NNODES);
  gemm_bias_kernel<<<G64x4, B, 0, stream>>>(m2_w, hmlp, m2_b, y2s, DFEAT, 2 * DFEAT, NNODES);
  // --- mlp (tar) ---
  copy_kernel<<<4096, B, 0, stream>>>(xt, cat, U);
  copy_kernel<<<4096, B, 0, stream>>>(dl1, cat + U, U);
  gemm_bias_kernel<<<G64x8, B, 0, stream>>>(m1_w, cat, m1_b, hmlp, 2 * DFEAT, 2 * DFEAT, NNODES);
  bn_relu_kernel<<<2 * DFEAT, B, 0, stream>>>(hmlp, m1_g, m1_be, NNODES);
  gemm_bias_kernel<<<G64x4, B, 0, stream>>>(m2_w, hmlp, m2_b, y2t, DFEAT, 2 * DFEAT, NNODES);

  // --- linear MMD ---
  final_mmd_kernel<<<1, B, 0, stream>>>(xs, y2s, xt, y2t, out, NNODES);
}
